// MultiHeadAttentionALiBi_1881195675705
// MI455X (gfx1250) — compile-verified, matrix operand order checked by reading, not run on hardware
//
#include <hip/hip_runtime.h>
#include <hip/hip_bf16.h>
#include <stdint.h>

// ---------------- CDNA5 WMMA plumbing (wave32) ----------------
typedef __attribute__((ext_vector_type(16))) __bf16 v16bf;
typedef __attribute__((ext_vector_type(8)))  float  v8f;
typedef __attribute__((ext_vector_type(4)))  int    v4i;
typedef unsigned short u16;

union FragAB {
    uint4 u[2];   // 32 bytes = 16 bf16 per lane
    v16bf v;
};

__device__ inline v8f v8f_zero() {
    v8f z = {0.f, 0.f, 0.f, 0.f, 0.f, 0.f, 0.f, 0.f};
    return z;
}

__device__ inline u16 f2bf(float f) {
    unsigned int u = __float_as_uint(f);
    u += 0x7FFFu + ((u >> 16) & 1u);   // round-to-nearest-even
    return (u16)(u >> 16);
}

__device__ inline v8f wmma_bf16(const FragAB& a, const FragAB& b, v8f c) {
    // D(16x16,f32) = A(16x32,bf16) x B(32x16,bf16) + C
    return __builtin_amdgcn_wmma_f32_16x16x32_bf16(
        /*neg_a=*/false, a.v, /*neg_b=*/false, b.v,
        /*c_mod=*/(short)0, c, /*reuse_a=*/false, /*reuse_b=*/false);
}

// ---------------- CDNA5 async global->LDS copy (ASYNCcnt-tracked) ----------------
// Builtin signature: void(v4i addrspace(1)* src, v4i addrspace(3)* dst, imm off, imm cpol)
typedef __attribute__((address_space(1))) v4i gv4i_t;
typedef __attribute__((address_space(3))) v4i lv4i_t;

#if __has_builtin(__builtin_amdgcn_global_load_async_to_lds_b128)
__device__ __forceinline__ void async_copy_b128(const u16* gsrc, u16* ldst) {
    __builtin_amdgcn_global_load_async_to_lds_b128(
        (gv4i_t*)gsrc, (lv4i_t*)ldst, 0, 0);
}
#else
__device__ __forceinline__ void async_copy_b128(const u16* gsrc, u16* ldst) {
    // VGLOBAL async encoding: VDST = per-lane LDS byte address, VADDR = 64-bit global addr
    unsigned loff = (unsigned)(size_t)(lv4i_t*)ldst;
    asm volatile("global_load_async_to_lds_b128 %0, %1, off"
                 :: "v"(loff), "v"(gsrc) : "memory");
}
#endif

#if __has_builtin(__builtin_amdgcn_s_wait_asynccnt)
#define WAIT_ASYNCCNT(n) __builtin_amdgcn_s_wait_asynccnt(n)
#else
#define WAIT_ASYNCCNT(n) asm volatile("s_wait_asynccnt %0" :: "i"(n) : "memory")
#endif

// ---------------- problem constants ----------------
#define BATCH 4
#define SEQ   2048
#define DIM   1024
#define HEADS 16
#define DHEAD 64
#define MROWS (BATCH * SEQ)   // 8192

// ---------------- fp32 -> bf16 conversion ----------------
__global__ void cvt_f32_bf16(const float* __restrict__ in, u16* __restrict__ out, int n) {
    int i = blockIdx.x * blockDim.x + threadIdx.x;
    if (i < n) out[i] = f2bf(in[i]);
}

// ---------------- tiled bf16 WMMA GEMM: C = A @ B^T + bias ----------------
// A: [M,K] bf16 row-major, B: [N,K] bf16 row-major (torch Linear weight).
// Async double-buffered LDS staging (GLOBAL_LOAD_ASYNC_TO_LDS_B128 + s_wait_asynccnt).
// MODE 0: store fp32 C[M,N].
// MODE 1: scatter qkv columns (n = h*192 + which*64 + d) into
//         q,k -> [B,H,S,DH] bf16 and v -> [B,H,DH,S] bf16 (pre-transposed).
template <int MODE>
__global__ __launch_bounds__(256) void gemm_bf16_wmma(
    const u16* __restrict__ A, const u16* __restrict__ Bw,
    const float* __restrict__ bias,
    float* __restrict__ Cf, u16* __restrict__ Qo, u16* __restrict__ Ko, u16* __restrict__ Vto,
    int M, int N, int K)
{
    __shared__ u16 As[2][128 * 32];   // 2 x 8 KB
    __shared__ u16 Bs[2][128 * 32];   // 2 x 8 KB

    const int tid  = threadIdx.x;
    const int wave = tid >> 5, lane = tid & 31;
    const int lm = lane & 15, lh = lane >> 4;
    const int wm = wave >> 2, wn = wave & 3;          // 2x4 wave grid
    const int m0 = blockIdx.y * 128, n0 = blockIdx.x * 128;

    const int lrow = tid >> 1;                        // 0..127
    const int lcol = (tid & 1) * 16;                  // 0 or 16 elements

    v8f acc[4][2];
#pragma unroll
    for (int mt = 0; mt < 4; mt++)
#pragma unroll
        for (int nt = 0; nt < 2; nt++) acc[mt][nt] = v8f_zero();

    // issue one stage of async tile copies (4 x B128 per thread)
    auto issue_stage = [&](int k0, int buf) {
        const u16* ga = A  + (size_t)(m0 + lrow) * K + k0 + lcol;
        const u16* gb = Bw + (size_t)(n0 + lrow) * K + k0 + lcol;
        async_copy_b128(ga,     &As[buf][lrow * 32 + lcol]);
        async_copy_b128(ga + 8, &As[buf][lrow * 32 + lcol + 8]);
        async_copy_b128(gb,     &Bs[buf][lrow * 32 + lcol]);
        async_copy_b128(gb + 8, &Bs[buf][lrow * 32 + lcol + 8]);
    };

    const int nst = K >> 5;          // K/32 stages
    issue_stage(0, 0);

    for (int t = 0; t < nst; ++t) {
        const int cur = t & 1;
        __syncthreads();             // all waves done reading buf cur^1 before refill
        if (t + 1 < nst) {
            issue_stage((t + 1) << 5, cur ^ 1);
            WAIT_ASYNCCNT(4);        // this wave's stage-t loads complete (in-order)
        } else {
            WAIT_ASYNCCNT(0);
        }
        __syncthreads();             // every wave's async data for buf cur landed

        const u16* as = As[cur];
        const u16* bs = Bs[cur];
        FragAB afr[4], bfr[2];
#pragma unroll
        for (int mt = 0; mt < 4; mt++) {
            int row = wm * 64 + mt * 16 + lm;
            afr[mt].u[0] = *(const uint4*)&as[row * 32 + 8 * lh];
            afr[mt].u[1] = *(const uint4*)&as[row * 32 + 16 + 8 * lh];
        }
#pragma unroll
        for (int nt = 0; nt < 2; nt++) {
            int row = wn * 32 + nt * 16 + lm;
            bfr[nt].u[0] = *(const uint4*)&bs[row * 32 + 8 * lh];
            bfr[nt].u[1] = *(const uint4*)&bs[row * 32 + 16 + 8 * lh];
        }
#pragma unroll
        for (int mt = 0; mt < 4; mt++)
#pragma unroll
            for (int nt = 0; nt < 2; nt++)
                acc[mt][nt] = wmma_bf16(afr[mt], bfr[nt], acc[mt][nt]);
    }

    // epilogue
#pragma unroll
    for (int mt = 0; mt < 4; mt++) {
#pragma unroll
        for (int nt = 0; nt < 2; nt++) {
            int col = n0 + wn * 32 + nt * 16 + lm;
            float bv = bias[col];
#pragma unroll
            for (int r = 0; r < 8; r++) {
                int grow = m0 + wm * 64 + mt * 16 + r + 8 * lh;
                float val = acc[mt][nt][r] + bv;
                if (MODE == 0) {
                    Cf[(size_t)grow * N + col] = val;
                } else {
                    int s  = grow & (SEQ - 1);
                    int bb = grow >> 11;
                    int hh    = col / 192;
                    int rem   = col - hh * 192;
                    int which = rem >> 6;
                    int d     = rem & 63;
                    size_t bh = (size_t)bb * HEADS + hh;
                    u16 bfv = f2bf(val);
                    if (which == 0)      Qo [(bh * SEQ + s) * DHEAD + d] = bfv;
                    else if (which == 1) Ko [(bh * SEQ + s) * DHEAD + d] = bfv;
                    else                 Vto[(bh * DHEAD + d) * SEQ + s] = bfv;
                }
            }
        }
    }
}

// ---------------- flash attention with ALiBi (bf16 WMMA, fp32 softmax) ----------------
// Block = (b, h, 128 query rows); 8 waves x 16 rows. 64-key K/V^T tiles staged with
// async double buffering; P restaged through per-wave LDS into A-fragment layout.
__global__ __launch_bounds__(256) void attn_alibi_wmma(
    const u16* __restrict__ Q, const u16* __restrict__ Kk, const u16* __restrict__ Vt,
    const float* __restrict__ alibi, u16* __restrict__ Out)
{
    __shared__ u16 Ks[2][64 * 64];    // 2 x 8 KB : K tile   [key][d]
    __shared__ u16 Vs[2][64 * 64];    // 2 x 8 KB : V^T tile [d][key]
    __shared__ u16 Ps[8][16 * 64];    // 16 KB    : per-wave probs in A-frag layout

    const int tid  = threadIdx.x;
    const int wave = tid >> 5, lane = tid & 31;
    const int lm = lane & 15, lh = lane >> 4;
    const int bh = blockIdx.y;                 // b*H + h
    const int hh = bh & (HEADS - 1);
    const int bb = bh >> 4;
    const int q0 = blockIdx.x * 128;
    const int qrow = q0 + wave * 16 + lm;      // this lane's A-fragment row

    // Q fragments for the whole kernel (DH=64 -> two K=32 chunks)
    FragAB qf[2];
#pragma unroll
    for (int kc = 0; kc < 2; kc++) {
        const u16* src = Q + ((size_t)bh * SEQ + qrow) * DHEAD + kc * 32;
        qf[kc].u[0] = *(const uint4*)(src + 8 * lh);
        qf[kc].u[1] = *(const uint4*)(src + 16 + 8 * lh);
    }

    v8f oacc[4];
    float rmax[8], rsum[8];
#pragma unroll
    for (int dt = 0; dt < 4; dt++) oacc[dt] = v8f_zero();
#pragma unroll
    for (int r = 0; r < 8; r++) { rmax[r] = -3.0e38f; rsum[r] = 0.f; }

    const int lrow = tid >> 2;            // 0..63
    const int lseg = (tid & 3) * 16;      // 0,16,32,48 elements

    // issue K/V^T tile stage (4 x B128 per thread)
    auto issue_tile = [&](int kb, int buf) {
        const u16* gk = Kk + ((size_t)bh * SEQ + kb + lrow) * DHEAD + lseg;
        const u16* gv = Vt + ((size_t)bh * DHEAD + lrow) * SEQ + kb + lseg;
        async_copy_b128(gk,     &Ks[buf][lrow * 64 + lseg]);
        async_copy_b128(gk + 8, &Ks[buf][lrow * 64 + lseg + 8]);
        async_copy_b128(gv,     &Vs[buf][lrow * 64 + lseg]);
        async_copy_b128(gv + 8, &Vs[buf][lrow * 64 + lseg + 8]);
    };

    const int ntiles = SEQ / 64;          // 32
    issue_tile(0, 0);

    for (int ti = 0; ti < ntiles; ++ti) {
        const int cur = ti & 1;
        const int kb  = ti * 64;
        __syncthreads();
        if (ti + 1 < ntiles) {
            issue_tile((ti + 1) * 64, cur ^ 1);
            WAIT_ASYNCCNT(4);
        } else {
            WAIT_ASYNCCNT(0);
        }
        __syncthreads();

        const u16* ks = Ks[cur];
        const u16* vs = Vs[cur];

        // scores S = Q K^T : 4 key sub-tiles x 2 K-chunks
        v8f sc[4];
#pragma unroll
        for (int nt = 0; nt < 4; nt++) {
            sc[nt] = v8f_zero();
#pragma unroll
            for (int kc = 0; kc < 2; kc++) {
                FragAB bf;
                const u16* bp = &ks[(nt * 16 + lm) * 64 + kc * 32];
                bf.u[0] = *(const uint4*)(bp + 8 * lh);
                bf.u[1] = *(const uint4*)(bp + 16 + 8 * lh);
                sc[nt] = wmma_bf16(qf[kc], bf, sc[nt]);
            }
        }

        // ALiBi (streaming, non-temporal) + online softmax
        float p[4][8];
#pragma unroll
        for (int nt = 0; nt < 4; nt++) {
#pragma unroll
            for (int r = 0; r < 8; r++) {
                int qi = q0 + wave * 16 + r + 8 * lh;
                int ki = kb + nt * 16 + lm;
                float al = __builtin_nontemporal_load(
                    &alibi[((size_t)hh * SEQ + qi) * SEQ + ki]);
                p[nt][r] = sc[nt][r] * 0.125f + al;   // 1/sqrt(64)
            }
        }
#pragma unroll
        for (int r = 0; r < 8; r++) {
            float mx = fmaxf(fmaxf(p[0][r], p[1][r]), fmaxf(p[2][r], p[3][r]));
#pragma unroll
            for (int o = 1; o < 16; o <<= 1) mx = fmaxf(mx, __shfl_xor(mx, o, 32));
            float mnew  = fmaxf(rmax[r], mx);
            float scale = __expf(rmax[r] - mnew);
            rmax[r] = mnew;
            float rs = 0.f;
#pragma unroll
            for (int nt = 0; nt < 4; nt++) {
                p[nt][r] = __expf(p[nt][r] - mnew);
                rs += p[nt][r];
            }
#pragma unroll
            for (int o = 1; o < 16; o <<= 1) rs += __shfl_xor(rs, o, 32);
            rsum[r] = rsum[r] * scale + rs;
#pragma unroll
            for (int dt = 0; dt < 4; dt++) oacc[dt][r] = oacc[dt][r] * scale;
        }

        // restage P through LDS into A-fragment layout (per-wave region, no barrier needed)
#pragma unroll
        for (int nt = 0; nt < 4; nt++)
#pragma unroll
            for (int r = 0; r < 8; r++)
                Ps[wave][(r + 8 * lh) * 64 + nt * 16 + lm] = f2bf(p[nt][r]);

        // O += P V : 4 d sub-tiles x 2 key-chunks
#pragma unroll
        for (int dt = 0; dt < 4; dt++) {
#pragma unroll
            for (int kc = 0; kc < 2; kc++) {
                FragAB af, bf;
                const u16* ap = &Ps[wave][lm * 64 + kc * 32];
                af.u[0] = *(const uint4*)(ap + 8 * lh);
                af.u[1] = *(const uint4*)(ap + 16 + 8 * lh);
                const u16* bp = &vs[(dt * 16 + lm) * 64 + kc * 32];
                bf.u[0] = *(const uint4*)(bp + 8 * lh);
                bf.u[1] = *(const uint4*)(bp + 16 + 8 * lh);
                oacc[dt] = wmma_bf16(af, bf, oacc[dt]);
            }
        }
    }

    // normalize and write attention output as bf16 [B*S, D] (feeds final GEMM)
#pragma unroll
    for (int dt = 0; dt < 4; dt++) {
#pragma unroll
        for (int r = 0; r < 8; r++) {
            int qi = q0 + wave * 16 + r + 8 * lh;
            float val = oacc[dt][r] / rsum[r];
            Out[((size_t)bb * SEQ + qi) * DIM + hh * DHEAD + dt * 16 + lm] = f2bf(val);
        }
    }
}

// ---------------- host-side orchestration ----------------
extern "C" void kernel_launch(void* const* d_in, const int* in_sizes, int n_in,
                              void* d_out, int out_size, void* d_ws, size_t ws_size,
                              hipStream_t stream) {
    const float* x     = (const float*)d_in[0];
    const float* alibi = (const float*)d_in[1];
    const float* w_qkv = (const float*)d_in[2];
    const float* b_qkv = (const float*)d_in[3];
    const float* w_o   = (const float*)d_in[4];
    const float* b_o   = (const float*)d_in[5];
    float* out = (float*)d_out;

    // workspace layout (88 MB total)
    char* ws = (char*)d_ws;
    u16* x_bf    = (u16*)(ws);                            // 8192*1024  = 16 MB
    u16* wqkv_bf = (u16*)(ws + (size_t)16 * 1024 * 1024); // 3072*1024 = 6 MB
    u16* wo_bf   = (u16*)(ws + (size_t)22 * 1024 * 1024); // 1024*1024 = 2 MB
    u16* qb      = (u16*)(ws + (size_t)24 * 1024 * 1024); // 16 MB
    u16* kb      = (u16*)(ws + (size_t)40 * 1024 * 1024); // 16 MB
    u16* vtb     = (u16*)(ws + (size_t)56 * 1024 * 1024); // 16 MB
    u16* attn_o  = (u16*)(ws + (size_t)72 * 1024 * 1024); // 16 MB

    const int n_x  = MROWS * DIM;        // 8388608
    const int n_wq = 3 * DIM * DIM;      // 3145728
    const int n_wo = DIM * DIM;          // 1048576
    cvt_f32_bf16<<<dim3((n_x  + 255) / 256), dim3(256), 0, stream>>>(x,     x_bf,    n_x);
    cvt_f32_bf16<<<dim3((n_wq + 255) / 256), dim3(256), 0, stream>>>(w_qkv, wqkv_bf, n_wq);
    cvt_f32_bf16<<<dim3((n_wo + 255) / 256), dim3(256), 0, stream>>>(w_o,   wo_bf,   n_wo);

    // QKV projection: [8192,1024] @ [3072,1024]^T, scatter epilogue into q/k/v^T
    gemm_bf16_wmma<1><<<dim3(3 * DIM / 128, MROWS / 128), dim3(256), 0, stream>>>(
        x_bf, wqkv_bf, b_qkv, nullptr, qb, kb, vtb, MROWS, 3 * DIM, DIM);

    // flash attention with ALiBi: grid = (query blocks, B*H)
    attn_alibi_wmma<<<dim3(SEQ / 128, BATCH * HEADS), dim3(256), 0, stream>>>(
        qb, kb, vtb, alibi, attn_o);

    // output projection: [8192,1024] @ [1024,1024]^T + b_o -> fp32 d_out
    gemm_bf16_wmma<0><<<dim3(DIM / 128, MROWS / 128), dim3(256), 0, stream>>>(
        attn_o, wo_bf, b_o, out, nullptr, nullptr, nullptr, MROWS, DIM, DIM);
}
